// Bottleneck_multi_60258391163149
// MI455X (gfx1250) — matrix operand order repaired and checked by reading, not compiled, not run on hardware
//
#include <hip/hip_runtime.h>
#include <hip/hip_bf16.h>
#include <cstddef>
#include <cstdint>

// ---------------------------------------------------------------------------
// Masked bottleneck block, bf16 WMMA pipeline for gfx1250 (wave32).
//   B=32, Cin=256, H=W=56 (HW=3136), width=64, Cout=256.
// h1/h2 intermediates are NHWC bf16 -> contiguous 128B/pixel, staged to LDS
// with global_load_async_to_lds_b128 (ASYNCcnt). All GEMMs through
// v_wmma_f32_16x16x32_bf16. Binary masks applied by row-redirect to a zero
// LDS row (no VALU multiply on the async-staged data).
// ---------------------------------------------------------------------------

typedef __bf16 bf16_t;
typedef bf16_t v16bf __attribute__((ext_vector_type(16)));
typedef float  v8f   __attribute__((ext_vector_type(8)));

#define WMMA_BF16(A, Bm, C) \
  __builtin_amdgcn_wmma_f32_16x16x32_bf16(false, (A), false, (Bm), (short)0, (C), false, false)

static constexpr int   BATCH = 32;
static constexpr int   CIN   = 256;
static constexpr int   HW    = 3136;   // 56*56
static constexpr int   IMG   = 56;
static constexpr int   WIDTH = 64;
static constexpr int   COUT  = 256;
static constexpr float EPS   = 1e-5f;

// LDS strides (bf16 elements); multiples of 8 -> 16B-aligned rows.
static constexpr int S1 = 264;   // conv1: K=256 + 8 pad
static constexpr int S2 = 584;   // conv2 weights: K=576 + 8 pad
static constexpr int SH = 72;    // conv2 halo / conv3: K=64 + 8 pad

// ---------------- async global->LDS 16B copy (CDNA5 VGLOBAL async ops) ------
// global_load_async_to_lds_b128: VDST = LDS byte address (addr[31:0] of flat
// LDS pointer), VADDR = 64-bit global address. Tracked by ASYNCcnt.
__device__ __forceinline__ void copy16_g2l(bf16_t* ldst, const bf16_t* gsrc) {
  asm volatile("global_load_async_to_lds_b128 %0, %1, off"
               :: "v"((uint32_t)(uintptr_t)ldst), "v"(gsrc)
               : "memory");
}
__device__ __forceinline__ void async_fence_lds() {
  asm volatile("s_wait_asynccnt 0x0" ::: "memory");
}

// Stage `rows` rows of `gstride` bf16 (16B-multiple) into LDS rows of
// `lstride` bf16, 256 threads cooperating, one b128 async per chunk.
__device__ __forceinline__ void stage_rows(bf16_t* lds, int lstride,
                                           const bf16_t* g, int gstride,
                                           int rows, int t) {
  int cpr = gstride >> 3;                    // 16B chunks per row
  int total = rows * cpr;
  for (int i = t; i < total; i += 256) {
    int r = i / cpr, ch = i - r * cpr;
    copy16_g2l(lds + r * lstride + ch * 8, g + (size_t)r * gstride + ch * 8);
  }
}

// ---------------- fragment loaders (wave32 layouts, CDNA5 ISA) ------
__device__ __forceinline__ v16bf frag_a(const bf16_t* __restrict__ base,
                                        int stride, int m0, int k0, int lane) {
  int r  = lane & 15;
  int kb = (lane >> 4) << 4;                 // 0 or 16
  const bf16_t* p = base + (m0 + r) * stride + k0 + kb;
  union { v16bf v; uint4 q[2]; } u;
  u.q[0] = *reinterpret_cast<const uint4*>(p);        // K = kb .. kb+7
  u.q[1] = *reinterpret_cast<const uint4*>(p + 8);    // K = kb+8 .. kb+15
  return u.v;
}

__device__ __forceinline__ v16bf frag_b(const bf16_t* __restrict__ base,
                                        int stride, int n0, int k0, int lane) {
  int n  = lane & 15;
  int kb = (lane >> 4) << 4;                 // 0 or 16
  const bf16_t* p = base + (n0 + n) * stride + k0 + kb;
  union { v16bf v; uint4 q[2]; } u;
  u.q[0] = *reinterpret_cast<const uint4*>(p);
  u.q[1] = *reinterpret_cast<const uint4*>(p + 8);
  return u.v;
}

// B fragment from an explicit (per-lane) LDS row of stride SH.
__device__ __forceinline__ v16bf frag_row(const bf16_t* __restrict__ base,
                                          int row, int k0, int lane) {
  int kb = (lane >> 4) << 4;
  const bf16_t* p = base + row * SH + k0 + kb;
  union { v16bf v; uint4 q[2]; } u;
  u.q[0] = *reinterpret_cast<const uint4*>(p);
  u.q[1] = *reinterpret_cast<const uint4*>(p + 8);
  return u.v;
}

// ---------------- kernel 1: weight conversion + folded BN constants ---------
__global__ void prep_kernel(const float* __restrict__ w1, const float* __restrict__ w2,
                            const float* __restrict__ w3,
                            const float* __restrict__ g1, const float* __restrict__ b1,
                            const float* __restrict__ m1, const float* __restrict__ v1,
                            const float* __restrict__ g2, const float* __restrict__ b2,
                            const float* __restrict__ m2, const float* __restrict__ v2,
                            const float* __restrict__ g3, const float* __restrict__ b3,
                            const float* __restrict__ m3, const float* __restrict__ v3,
                            bf16_t* __restrict__ w1b, bf16_t* __restrict__ w2b,
                            bf16_t* __restrict__ w3b, float* __restrict__ bnc) {
  int t  = blockIdx.x * blockDim.x + threadIdx.x;
  int nt = gridDim.x * blockDim.x;
  for (int i = t; i < WIDTH * CIN; i += nt)  w1b[i] = (bf16_t)w1[i];
  for (int i = t; i < COUT * WIDTH; i += nt) w3b[i] = (bf16_t)w3[i];
  for (int i = t; i < WIDTH * WIDTH * 9; i += nt) {     // OIHW -> [co][tap*64+ci]
    int co = i / 576, rest = i % 576, tap = rest / 64, ci = rest % 64;
    w2b[i] = (bf16_t)w2[(co * 64 + ci) * 9 + tap];
  }
  for (int i = t; i < 64; i += nt) {
    float s = g1[i] * rsqrtf(v1[i] + EPS);
    bnc[i] = s; bnc[64 + i] = b1[i] - m1[i] * s;
  }
  for (int i = t; i < 64; i += nt) {
    float s = g2[i] * rsqrtf(v2[i] + EPS);
    bnc[128 + i] = s; bnc[192 + i] = b2[i] - m2[i] * s;
  }
  for (int i = t; i < 256; i += nt) {
    float s = g3[i] * rsqrtf(v3[i] + EPS);
    bnc[256 + i] = s; bnc[512 + i] = b3[i] - m3[i] * s;
  }
}

// ---------------- kernel 2: conv1 (1x1, 256->64) + BN1/ReLU + mask logit ----
__global__ __launch_bounds__(256) void conv1_kernel(
    const float* __restrict__ x, const bf16_t* __restrict__ w1b,
    const float* __restrict__ bnc, const float* __restrict__ mask_w,
    const float* __restrict__ mask_b, bf16_t* __restrict__ h1,
    float* __restrict__ mask) {
  extern __shared__ char smem[];
  bf16_t* Wl  = (bf16_t*)smem;                 // 64 x S1
  bf16_t* Xl  = Wl + 64 * S1;                  // 64 x S1 (pixel-major [n][k])
  float*  par = (float*)(Xl + 64 * S1);        // 4 x 64 partial mask dots
  float*  bns = par + 256;                     // 128: scale1 | shift1

  int t   = threadIdx.x;
  int b   = blockIdx.y;
  int hw0 = blockIdx.x * 64;

  stage_rows(Wl, S1, w1b, CIN, 64, t);         // async weights -> LDS
  if (t < 128) bns[t] = bnc[t];

  // X tile: f32 -> bf16 conversion (VALU path), pixel p coalesced across lanes.
  int p = t & 63, cq = t >> 6;
  const float* xg = x + (size_t)b * CIN * HW + hw0 + p;
  __builtin_prefetch(xg + (size_t)(cq * 64) * HW, 0, 1);
  float acc = 0.f;
  for (int j = 0; j < 64; ++j) {
    int   c  = cq * 64 + j;
    float xv = xg[(size_t)c * HW];
    Xl[p * S1 + c] = (bf16_t)xv;
    acc += xv * mask_w[c];
  }
  par[cq * 64 + p] = acc;
  async_fence_lds();
  __syncthreads();

  if (t < 64) {
    float soft = par[t] + par[64 + t] + par[128 + t] + par[192 + t] + mask_b[0];
    mask[(size_t)b * HW + hw0 + t] = (soft >= 0.f) ? 1.f : 0.f;
  }

  int lane = t & 31, wid = t >> 5;
  int wm = wid >> 1, wn = wid & 1, m0 = wm * 16;
  v8f c0 = {}, c1 = {};
  for (int k0 = 0; k0 < CIN; k0 += 32) {
    v16bf a  = frag_a(Wl, S1, m0, k0, lane);
    v16bf bL = frag_b(Xl, S1, wn * 32,      k0, lane);
    v16bf bR = frag_b(Xl, S1, wn * 32 + 16, k0, lane);
    c0 = WMMA_BF16(a, bL, c0);
    c1 = WMMA_BF16(a, bR, c1);
  }

  // Epilogue: BN1+ReLU; h1 is NHWC -> one 16B store per accumulator.
  int ncol = lane & 15, mo = (lane >> 4) << 3;
  v8f cc[2] = {c0, c1};
  for (int s = 0; s < 2; ++s) {
    int n = wn * 32 + s * 16 + ncol;
    union { uint4 q; bf16_t e[8]; } o;
    for (int r = 0; r < 8; ++r) {
      int   m = m0 + mo + r;
      float v = cc[s][r] * bns[m] + bns[64 + m];
      o.e[r] = (bf16_t)(v > 0.f ? v : 0.f);
    }
    *reinterpret_cast<uint4*>(&h1[((size_t)b * HW + hw0 + n) * WIDTH + m0 + mo]) = o.q;
  }
}

// ---------------- kernel 3: 3x3 binary dilation of the mask -----------------
__global__ void dilate_kernel(const float* __restrict__ mask,
                              float* __restrict__ maskd) {
  int i = blockIdx.x * 256 + threadIdx.x;
  if (i >= BATCH * HW) return;
  int b = i / HW, hw = i % HW, y = hw / IMG, xx = hw % IMG;
  float m = 0.f;
  for (int dy = -1; dy <= 1; ++dy)
    for (int dx = -1; dx <= 1; ++dx) {
      int yy = y + dy, xc = xx + dx;
      if (yy >= 0 && yy < IMG && xc >= 0 && xc < IMG)
        m = fmaxf(m, mask[(size_t)b * HW + yy * IMG + xc]);
    }
  maskd[i] = m;
}

// ---------------- kernel 4: conv2 (3x3, 64->64) + BN2/ReLU + masks ----------
// 10x10 halo of raw NHWC h1 async-staged; mask_dilate/zero-pad applied by
// redirecting the B-fragment row to a zero LDS row (mask is binary).
__global__ __launch_bounds__(256) void conv2_kernel(
    const bf16_t* __restrict__ h1, const bf16_t* __restrict__ w2b,
    const float* __restrict__ bnc, const float* __restrict__ mask,
    const float* __restrict__ maskd, bf16_t* __restrict__ h2) {
  extern __shared__ char smem[];
  bf16_t* Wl  = (bf16_t*)smem;                 // 64 x S2 ([co][tap*64+ci])
  bf16_t* Xh  = Wl + 64 * S2;                  // 101 x SH (row 100 = zeros)
  float*  bns = (float*)(Xh + 101 * SH);       // 128: scale2 | shift2
  float*  mdl = bns + 128;                     // 100: maskd halo tile

  int t  = threadIdx.x;
  int b  = blockIdx.y;
  int ty = blockIdx.x / 7, tx = blockIdx.x % 7;

  stage_rows(Wl, S2, w2b, 576, 64, t);         // async weights -> LDS
  if (t < 128) bns[t] = bnc[128 + t];
  if (t < SH / 8) {                            // zero redirect row
    uint4 z = {};
    *reinterpret_cast<uint4*>(&Xh[100 * SH + t * 8]) = z;
  }

  // Halo: 100 pixels x 128B raw h1 rows, async b128 (OOB rows never read).
  const bf16_t* hb = h1 + (size_t)b * HW * WIDTH;
  for (int i = t; i < 100 * 8; i += 256) {
    int p = i >> 3, ch = i & 7;
    int hy = p / 10, hx = p % 10;
    int gy = ty * 8 + hy - 1, gx = tx * 8 + hx - 1;
    if (gy >= 0 && gy < IMG && gx >= 0 && gx < IMG)
      copy16_g2l(&Xh[p * SH + ch * 8],
                 hb + (size_t)(gy * IMG + gx) * WIDTH + ch * 8);
  }
  if (t < 100) {
    int hy = t / 10, hx = t % 10;
    int gy = ty * 8 + hy - 1, gx = tx * 8 + hx - 1;
    bool ok = (gy >= 0 && gy < IMG && gx >= 0 && gx < IMG);
    mdl[t] = ok ? maskd[(size_t)b * HW + gy * IMG + gx] : 0.f;
  }
  async_fence_lds();
  __syncthreads();

  int lane = t & 31, wid = t >> 5;
  int wm = wid >> 1, wn = wid & 1, m0 = wm * 16;
  v8f c0 = {}, c1 = {};
  for (int tap = 0; tap < 9; ++tap) {
    int dy = tap / 3, dx = tap % 3;            // halo origin (-1,-1) folded in
    int pi0 = wn * 32 + (lane & 15);
    int pi1 = pi0 + 16;
    int hp0 = ((pi0 >> 3) + dy) * 10 + ((pi0 & 7) + dx);
    int hp1 = ((pi1 >> 3) + dy) * 10 + ((pi1 & 7) + dx);
    hp0 = (mdl[hp0] != 0.f) ? hp0 : 100;       // masked/pad -> zero row
    hp1 = (mdl[hp1] != 0.f) ? hp1 : 100;
    for (int k0 = 0; k0 < WIDTH; k0 += 32) {
      v16bf a = frag_a(Wl, S2, m0, tap * 64 + k0, lane);
      c0 = WMMA_BF16(a, frag_row(Xh, hp0, k0, lane), c0);
      c1 = WMMA_BF16(a, frag_row(Xh, hp1, k0, lane), c1);
    }
  }

  int ncol = lane & 15, mo = (lane >> 4) << 3;
  v8f cc[2] = {c0, c1};
  for (int s = 0; s < 2; ++s) {
    int pi = wn * 32 + s * 16 + ncol;
    int gofs = (ty * 8 + (pi >> 3)) * IMG + (tx * 8 + (pi & 7));
    float mk = mask[(size_t)b * HW + gofs];
    union { uint4 q; bf16_t e[8]; } o;
    for (int r = 0; r < 8; ++r) {
      int   m = m0 + mo + r;
      float v = cc[s][r] * bns[m] + bns[64 + m];
      o.e[r] = (bf16_t)((v > 0.f ? v : 0.f) * mk);
    }
    *reinterpret_cast<uint4*>(&h2[((size_t)b * HW + gofs) * WIDTH + m0 + mo]) = o.q;
  }
}

// ---------------- kernel 5: conv3 (1x1, 64->256) + BN3*mask + residual ------
__global__ __launch_bounds__(256) void conv3_kernel(
    const bf16_t* __restrict__ h2, const bf16_t* __restrict__ w3b,
    const float* __restrict__ bnc, const float* __restrict__ mask,
    const float* __restrict__ x, float* __restrict__ out) {
  extern __shared__ char smem[];
  bf16_t* Wl  = (bf16_t*)smem;                 // 64 x SH
  bf16_t* Xl  = Wl + 64 * SH;                  // 64 x SH
  float*  bns = (float*)(Xl + 64 * SH);        // 128: scale3|shift3 slice

  int t   = threadIdx.x;
  int b   = blockIdx.y;
  int mg  = blockIdx.z;                        // output channel group 0..3
  int hw0 = blockIdx.x * 64;
  int cobase = mg * 64;

  stage_rows(Wl, SH, w3b + (size_t)cobase * 64, 64, 64, t);      // async
  stage_rows(Xl, SH, h2 + ((size_t)b * HW + hw0) * WIDTH, 64, 64, t);
  if (t < 64)       bns[t] = bnc[256 + cobase + t];
  else if (t < 128) bns[t] = bnc[512 + cobase + (t - 64)];
  async_fence_lds();
  __syncthreads();

  int lane = t & 31, wid = t >> 5;
  int wm = wid >> 1, wn = wid & 1, m0 = wm * 16;
  v8f c0 = {}, c1 = {};
  for (int k0 = 0; k0 < WIDTH; k0 += 32) {
    v16bf a  = frag_a(Wl, SH, m0, k0, lane);
    v16bf bL = frag_b(Xl, SH, wn * 32,      k0, lane);
    v16bf bR = frag_b(Xl, SH, wn * 32 + 16, k0, lane);
    c0 = WMMA_BF16(a, bL, c0);
    c1 = WMMA_BF16(a, bR, c1);
  }

  int ncol = lane & 15, mo = (lane >> 4) << 3;
  v8f cc[2] = {c0, c1};
  for (int s = 0; s < 2; ++s) {
    int n  = wn * 32 + s * 16 + ncol;
    int hw = hw0 + n;
    float mk = mask[(size_t)b * HW + hw];
    for (int r = 0; r < 8; ++r) {
      int   ml = m0 + mo + r;                  // channel within group
      int   m  = cobase + ml;
      float v  = cc[s][r] * bns[ml] + bns[64 + ml];
      size_t gi = (size_t)(b * COUT + m) * HW + hw;   // NCHW f32 output
      float res = x[gi] + v * mk;
      out[gi] = res > 0.f ? res : 0.f;
    }
  }
}

// ---------------------------------------------------------------------------
extern "C" void kernel_launch(void* const* d_in, const int* in_sizes, int n_in,
                              void* d_out, int out_size, void* d_ws, size_t ws_size,
                              hipStream_t stream) {
  const float* x   = (const float*)d_in[0];
  const float* w1  = (const float*)d_in[1];
  const float* w2  = (const float*)d_in[2];
  const float* w3  = (const float*)d_in[3];
  const float* g1  = (const float*)d_in[4];
  const float* b1  = (const float*)d_in[5];
  const float* m1  = (const float*)d_in[6];
  const float* v1  = (const float*)d_in[7];
  const float* g2  = (const float*)d_in[8];
  const float* b2  = (const float*)d_in[9];
  const float* m2  = (const float*)d_in[10];
  const float* v2  = (const float*)d_in[11];
  const float* g3  = (const float*)d_in[12];
  const float* b3  = (const float*)d_in[13];
  const float* m3  = (const float*)d_in[14];
  const float* v3  = (const float*)d_in[15];
  const float* mkw = (const float*)d_in[16];
  const float* mkb = (const float*)d_in[17];

  // Workspace layout (bytes), ~26.6 MB total, 256B aligned segments.
  char* ws = (char*)d_ws;
  bf16_t* w1b   = (bf16_t*)(ws);                         //  32768
  bf16_t* w2b   = (bf16_t*)(ws + 32768);                 //  73728
  bf16_t* w3b   = (bf16_t*)(ws + 106496);                //  32768
  float*  bnc   = (float*) (ws + 139264);                //   3072
  float*  mask  = (float*) (ws + 142336);                // 401408
  float*  maskd = (float*) (ws + 543744);                // 401408
  bf16_t* h1    = (bf16_t*)(ws + 945152);                // 12845056 (NHWC)
  bf16_t* h2    = (bf16_t*)(ws + 13790208);              // 12845056 (NHWC)
  float*  out   = (float*)d_out;

  prep_kernel<<<32, 256, 0, stream>>>(w1, w2, w3, g1, b1, m1, v1,
                                      g2, b2, m2, v2, g3, b3, m3, v3,
                                      w1b, w2b, w3b, bnc);

  size_t lds1 = (size_t)(64 * S1 * 2) * 2 + (256 + 128) * 4;             // 69120
  conv1_kernel<<<dim3(49, 32), 256, lds1, stream>>>(x, w1b, bnc, mkw, mkb,
                                                    h1, mask);

  dilate_kernel<<<(BATCH * HW + 255) / 256, 256, 0, stream>>>(mask, maskd);

  size_t lds2 = (size_t)(64 * S2 + 101 * SH) * 2 + (128 + 100) * 4;      // 90208
  conv2_kernel<<<dim3(49, 32), 256, lds2, stream>>>(h1, w2b, bnc, mask, maskd,
                                                    h2);

  size_t lds3 = (size_t)(64 * SH * 2) * 2 + 128 * 4;                     // 18944
  conv3_kernel<<<dim3(49, 32, 4), 256, lds3, stream>>>(h2, w3b, bnc, mask, x,
                                                       out);
}
